// MoEGraphProjector_42099269436306
// MI455X (gfx1250) — matrix operand order repaired and checked by reading, not compiled, not run on hardware
//
#include <hip/hip_runtime.h>
#include <math.h>

typedef __bf16 bf16_t;
typedef __attribute__((ext_vector_type(16))) __bf16 v16bf;
typedef __attribute__((ext_vector_type(8)))  __bf16 v8bf;
typedef __attribute__((ext_vector_type(8)))  float  v8f;

#define DMM   2432
#define DLLM  4096
#define BATCH 256
#define SEQ   64
#define NEXP  4
#define TOPK  2

#define BSTRIDE 40   // padded LDS stride (bf16 elems): 80B -> 20 banks, conflict-free
#define NCOLS   256  // output cols per workgroup

// ---------------------------------------------------------------------------
// Kernel 1: routing — logits, softmax, top-2, weights, aux_loss.
// One block of 256 threads; thread b handles batch row b.
// ---------------------------------------------------------------------------
__global__ __launch_bounds__(256) void routing_kernel(
    const float* __restrict__ rf,      // (B, DMM)
    const float* __restrict__ gate_w,  // (E, DMM)
    int*   __restrict__ ws_idx,        // (B, 2)
    float* __restrict__ ws_w,          // (B, 2)
    float* __restrict__ aux_out)       // scalar
{
    __shared__ float s_cnt[NEXP];
    __shared__ float s_p[NEXP];
    const int b = threadIdx.x;
    if (b < NEXP) { s_cnt[b] = 0.0f; s_p[b] = 0.0f; }
    __syncthreads();

    // logits[e] = rf[b] . gate_w[e]
    float l0 = 0.f, l1 = 0.f, l2 = 0.f, l3 = 0.f;
    const float4* x4 = (const float4*)(rf + (size_t)b * DMM);
    const float4* g0 = (const float4*)(gate_w + 0 * DMM);
    const float4* g1 = (const float4*)(gate_w + 1 * DMM);
    const float4* g2 = (const float4*)(gate_w + 2 * DMM);
    const float4* g3 = (const float4*)(gate_w + 3 * DMM);
    for (int i = 0; i < DMM / 4; ++i) {
        float4 x = x4[i];
        float4 a = g0[i], c = g1[i], d = g2[i], e = g3[i];
        l0 += x.x*a.x + x.y*a.y + x.z*a.z + x.w*a.w;
        l1 += x.x*c.x + x.y*c.y + x.z*c.z + x.w*c.w;
        l2 += x.x*d.x + x.y*d.y + x.z*d.z + x.w*d.w;
        l3 += x.x*e.x + x.y*e.y + x.z*e.z + x.w*e.w;
    }
    float l[NEXP] = { l0, l1, l2, l3 };

    // full softmax (for P = mean of probs)
    float mx = l[0];
    for (int e = 1; e < NEXP; ++e) mx = fmaxf(mx, l[e]);
    float p[NEXP], psum = 0.f;
    for (int e = 0; e < NEXP; ++e) { p[e] = __expf(l[e] - mx); psum += p[e]; }
    float inv = 1.0f / psum;
    for (int e = 0; e < NEXP; ++e) p[e] *= inv;

    // top-2 (ties -> lowest index, matching jax.lax.top_k)
    int i0 = 0;
    for (int e = 1; e < NEXP; ++e) if (l[e] > l[i0]) i0 = e;
    int i1 = (i0 == 0) ? 1 : 0;
    for (int e = 0; e < NEXP; ++e) if (e != i0 && l[e] > l[i1]) i1 = e;

    // softmax over the 2 selected logits (l[i0] >= l[i1])
    float e1 = __expf(l[i1] - l[i0]);
    float w0 = 1.0f / (1.0f + e1);
    float w1 = e1 * w0;

    ws_idx[2 * b + 0] = i0;
    ws_idx[2 * b + 1] = i1;
    ws_w[2 * b + 0] = w0;
    ws_w[2 * b + 1] = w1;

    atomicAdd(&s_cnt[i0], 1.0f);
    atomicAdd(&s_cnt[i1], 1.0f);
    for (int e = 0; e < NEXP; ++e) atomicAdd(&s_p[e], p[e]);
    __syncthreads();

    if (b == 0) {
        float aux = 0.f;
        for (int e = 0; e < NEXP; ++e) {
            float f = s_cnt[e] / (float)(BATCH * TOPK);
            float P = s_p[e] / (float)BATCH;
            aux += f * P;
        }
        aux_out[0] = (float)NEXP * aux;
    }
}

// ---------------------------------------------------------------------------
// Kernel 2: fused top-2 expert GEMM + bias + weighted combine + mask.
// Grid: (DLLM/256, B). Block: 256 threads = 8 waves.
// Workgroup tile: 64 rows x 256 cols.  Wave w owns a 32-col slab across ALL
// 64 rows: 4 A fragments + 4 B fragments all preloaded into distinct
// registers each K-step so all 16 ds_load_b128 are in flight before the
// 16-WMMA chain starts (partial s_wait_dscnt overlap).  Routing weight is
// folded into the bf16 conversion of each expert's B tile -> both experts
// accumulate into the same f32 accumulators in one K loop.
// ---------------------------------------------------------------------------
__global__ __launch_bounds__(256) void moe_gemm_kernel(
    const float* __restrict__ X,       // (B, S, DMM)
    const float* __restrict__ W,       // (E, DLLM, DMM)
    const float* __restrict__ bias,    // (E, DLLM)
    const unsigned char* __restrict__ mask, // (B, S) bool
    const int*   __restrict__ ws_idx,
    const float* __restrict__ ws_w,
    float* __restrict__ out)           // (B, S, DLLM)
{
    __shared__ __align__(16) bf16_t lds_a[SEQ * BSTRIDE];        // 64x32 (pad 40), 5 KB
    __shared__ __align__(16) bf16_t lds_b[2][NCOLS * BSTRIDE];   // 2 x 256x32 (pad 40), 40 KB

    const int b      = blockIdx.y;
    const int ocol0  = blockIdx.x * NCOLS;
    const int tid    = threadIdx.x;
    const int lane   = tid & 31;
    const int wave   = tid >> 5;       // wave w -> cols [w*32, w*32+32)
    const int lane16 = lane & 15;
    const int hsel   = lane >> 4;      // lane half (K-chunk select)

    const int   e0 = ws_idx[2 * b + 0];
    const int   e1 = ws_idx[2 * b + 1];
    const float w0 = ws_w[2 * b + 0];
    const float w1 = ws_w[2 * b + 1];

    const float* Xb  = X + (size_t)b * SEQ * DMM;
    const float* We0 = W + (size_t)e0 * DLLM * DMM;
    const float* We1 = W + (size_t)e1 * DLLM * DMM;

    v8f acc[4][2];                     // [row strip][col tile]
    #pragma unroll
    for (int r = 0; r < 4; ++r)
        #pragma unroll
        for (int c = 0; c < 2; ++c)
            #pragma unroll
            for (int j = 0; j < 8; ++j) acc[r][c][j] = 0.0f;

    // cooperative-load index split
    const int a_s  = tid >> 2;         // 0..63  (A row), 8 floats each
    const int a_q  = tid & 3;
    const int bo_r = tid >> 3;         // 0..31  (B row within pass)
    const int bo_q = tid & 7;          // float4 index within 32-float row seg

    for (int d0 = 0; d0 < DMM; d0 += 32) {
        __syncthreads();
        // --- stage A tile (64x32 fp32 -> bf16) ---
        {
            const float4* src = (const float4*)(Xb + (size_t)a_s * DMM + d0 + a_q * 8);
            float4 p0 = src[0], p1 = src[1];
            bf16_t* dst = &lds_a[a_s * BSTRIDE + a_q * 8];
            dst[0] = (bf16_t)p0.x; dst[1] = (bf16_t)p0.y;
            dst[2] = (bf16_t)p0.z; dst[3] = (bf16_t)p0.w;
            dst[4] = (bf16_t)p1.x; dst[5] = (bf16_t)p1.y;
            dst[6] = (bf16_t)p1.z; dst[7] = (bf16_t)p1.w;
        }
        // --- stage B tiles for both experts (32x256 fp32 -> bf16, pre-scaled) ---
        #pragma unroll
        for (int pass = 0; pass < 8; ++pass) {
            const int o = pass * 32 + bo_r;                  // 0..255
            const size_t roff = (size_t)(ocol0 + o) * DMM + d0;
            const float4* s0 = (const float4*)(We0 + roff) + bo_q;
            const float4* s1 = (const float4*)(We1 + roff) + bo_q;
            // prefetch next K-step's cacheline for this row
            __builtin_prefetch((const void*)(We0 + roff + 32), 0, 1);
            __builtin_prefetch((const void*)(We1 + roff + 32), 0, 1);
            float4 p = *s0;
            bf16_t* dst = &lds_b[0][o * BSTRIDE + bo_q * 4];
            dst[0] = (bf16_t)(w0 * p.x); dst[1] = (bf16_t)(w0 * p.y);
            dst[2] = (bf16_t)(w0 * p.z); dst[3] = (bf16_t)(w0 * p.w);
            float4 q = *s1;
            bf16_t* dst1 = &lds_b[1][o * BSTRIDE + bo_q * 4];
            dst1[0] = (bf16_t)(w1 * q.x); dst1[1] = (bf16_t)(w1 * q.y);
            dst1[2] = (bf16_t)(w1 * q.z); dst1[3] = (bf16_t)(w1 * q.w);
        }
        __syncthreads();

        // --- preload ALL fragments: 4 A (row strips) + 4 B (2 tiles x 2
        //     experts) so all 16 ds_load_b128 are in flight together ---
        // A layout: lane row = r*16 + lane%16, K = {kb..kb+7, kb+16..kb+23},
        // kb = 8*(lane/16)
        const int kb = hsel * 8;
        v16bf afrag[4];
        #pragma unroll
        for (int r = 0; r < 4; ++r) {
            const int arow = r * 16 + lane16;
            v8bf alo = *(const v8bf*)&lds_a[arow * BSTRIDE + kb];
            v8bf ahi = *(const v8bf*)&lds_a[arow * BSTRIDE + kb + 16];
            #pragma unroll
            for (int i = 0; i < 8; ++i) { afrag[r][i] = alo[i]; afrag[r][8 + i] = ahi[i]; }
        }
        // B layout: lane col = lane%16, K = {kb..kb+7, kb+16..kb+23}, kb = 8*(lane/16)
        v16bf bfrag[4];                 // [kk*2 + c]
        #pragma unroll
        for (int kk = 0; kk < 2; ++kk) {
            #pragma unroll
            for (int c = 0; c < 2; ++c) {
                const int ob = wave * 32 + c * 16 + lane16;
                const bf16_t* bp = &lds_b[kk][ob * BSTRIDE + hsel * 8];
                v8bf blo = *(const v8bf*)bp;
                v8bf bhi = *(const v8bf*)(bp + 16);
                #pragma unroll
                for (int i = 0; i < 8; ++i) {
                    bfrag[kk * 2 + c][i] = blo[i];
                    bfrag[kk * 2 + c][8 + i] = bhi[i];
                }
            }
        }

        // --- 16 WMMAs, consuming B fragments in load order ---
        #pragma unroll
        for (int kk = 0; kk < 2; ++kk) {
            #pragma unroll
            for (int c = 0; c < 2; ++c) {
                #pragma unroll
                for (int r = 0; r < 4; ++r) {
                    acc[r][c] = __builtin_amdgcn_wmma_f32_16x16x32_bf16(
                        false, afrag[r], false, bfrag[kk * 2 + c],
                        (short)0, acc[r][c], false, false);
                }
            }
        }
    }

    // --- epilogue: + weighted bias, apply mask, store ---
    const unsigned char* mb = mask + (size_t)b * SEQ;
    #pragma unroll
    for (int c = 0; c < 2; ++c) {
        const int o = ocol0 + wave * 32 + c * 16 + lane16;
        const float bi = w0 * bias[e0 * DLLM + o] + w1 * bias[e1 * DLLM + o];
        #pragma unroll
        for (int r = 0; r < 4; ++r) {
            #pragma unroll
            for (int j = 0; j < 8; ++j) {
                const int s = r * 16 + j + hsel * 8;   // C layout: M = j + 8*(lane/16)
                float v = acc[r][c][j] + bi;
                v = mb[s] ? v : 0.0f;
                out[((size_t)b * SEQ + s) * DLLM + o] = v;
            }
        }
    }
}

// ---------------------------------------------------------------------------
extern "C" void kernel_launch(void* const* d_in, const int* in_sizes, int n_in,
                              void* d_out, int out_size, void* d_ws, size_t ws_size,
                              hipStream_t stream) {
    (void)in_sizes; (void)n_in; (void)out_size; (void)ws_size;
    const float* graph_emb = (const float*)d_in[0];  // (B,S,DMM)
    const float* rf        = (const float*)d_in[1];  // (B,DMM)
    const float* gate_w    = (const float*)d_in[2];  // (E,DMM)
    const float* expert_w  = (const float*)d_in[3];  // (E,DLLM,DMM)
    const float* expert_b  = (const float*)d_in[4];  // (E,DLLM)
    const unsigned char* gmask = (const unsigned char*)d_in[5]; // (B,S) bool

    float* out = (float*)d_out;                      // combined (B,S,DLLM) ++ aux
    float* aux_out = out + (size_t)BATCH * SEQ * DLLM;

    int*   ws_idx = (int*)d_ws;                      // B*2 ints
    float* ws_w   = (float*)((char*)d_ws + BATCH * 2 * sizeof(int));

    routing_kernel<<<1, 256, 0, stream>>>(rf, gate_w, ws_idx, ws_w, aux_out);

    dim3 grid(DLLM / NCOLS, BATCH);
    moe_gemm_kernel<<<grid, 256, 0, stream>>>(graph_emb, expert_w, expert_b,
                                              gmask, ws_idx, ws_w, out);
}
